// MultiHeadAttention_34626026340719
// MI455X (gfx1250) — matrix operand order repaired and checked by reading, not compiled, not run on hardware
//
#include <hip/hip_runtime.h>

typedef __attribute__((ext_vector_type(16))) _Float16 v16h;
typedef __attribute__((ext_vector_type(8)))  float    v8f;
typedef __attribute__((ext_vector_type(4)))  int      v4i;

#define EMBED   1024
#define HEADS   16
#define HDIM    64
#define SEQ     2048
#define BATCH   2
#define MROWS   (BATCH * SEQ)   // 4096
#define HALF_WIN 128

#define AS1 __attribute__((address_space(1)))
#define AS3 __attribute__((address_space(3)))

union FragA {
    v16h v;
    uint4 q[2];
    _Float16 h[16];
};

static __device__ __forceinline__ v8f wmma_f16(const FragA& a, const FragA& b, v8f c) {
    return __builtin_amdgcn_wmma_f32_16x16x32_f16(
        /*neg_a=*/false, a.v, /*neg_b=*/false, b.v,
        /*c_mod=*/(short)0, c, /*reuse_a=*/false, /*reuse_b=*/false);
}

// Wait for all outstanding async global<->LDS transfers (ASYNCcnt == 0).
static __device__ __forceinline__ void wait_async0() {
#if __has_builtin(__builtin_amdgcn_s_wait_asynccnt)
    __builtin_amdgcn_s_wait_asynccnt(0);
#else
    asm volatile("s_wait_asynccnt 0x0" ::: "memory");
#endif
}

// 16B async copy global -> LDS (falls back to sync copy if builtin missing).
static __device__ __forceinline__ void async_b128(const _Float16* gsrc, _Float16* ldst) {
#if __has_builtin(__builtin_amdgcn_global_load_async_to_lds_b128)
    __builtin_amdgcn_global_load_async_to_lds_b128(
        (AS1 v4i*)gsrc, (AS3 v4i*)ldst, 0, 0);
#else
    *(uint4*)ldst = *(const uint4*)gsrc;
#endif
}

// ---------------------------------------------------------------------------
// fp32 -> f16 conversion, 4 elements per thread
// ---------------------------------------------------------------------------
__global__ void cvt_f32_to_f16(const float4* __restrict__ in,
                               uint2* __restrict__ out, int n4) {
    int i = blockIdx.x * blockDim.x + threadIdx.x;
    if (i < n4) {
        float4 v = in[i];
        union { _Float16 h[4]; uint2 u; } p;
        p.h[0] = (_Float16)v.x;
        p.h[1] = (_Float16)v.y;
        p.h[2] = (_Float16)v.z;
        p.h[3] = (_Float16)v.w;
        out[i] = p.u;
    }
}

// ---------------------------------------------------------------------------
// Tiled WMMA GEMM:  Y[m,n] = sum_k X[m,k] * W[n,k] + bias[n]
// X: (M=4096, K=1024) f16 row-major; W: (N=1024, K=1024) f16 row-major.
// Block = 256 threads = 8 waves, each computing 32(M) x 64(N): block 256x64.
// W tile (64x32) double-buffered in LDS via async global->LDS loads,
// stride 40 rows (bank-conflict-free b128 reads). 8 WMMAs per k-step/wave.
// mode 0: f16 scatter to (b*H+h, s, d)      [Q, K]
// mode 1: f16 scatter to (b*H+h, d, s)      [V transposed for attention]
// mode 2: f32 row-major (M, N)              [final output]
// ---------------------------------------------------------------------------
__global__ __launch_bounds__(256) void gemm_wmma(
    const _Float16* __restrict__ X,
    const _Float16* __restrict__ W,
    const float* __restrict__ bias,
    void* __restrict__ Y, int mode) {
    __shared__ _Float16 ldsW[2][64 * 40];

    const int tid  = threadIdx.x;
    const int wave = tid >> 5;
    const int lane = tid & 31;
    const int ln   = lane & 15;
    const int hi   = lane >> 4;

    const int m0 = blockIdx.x * 256 + wave * 32;
    const int n0 = blockIdx.y * 64;

    // cooperative W-tile mapping: 64 rows x 32 cols, 16B per thread
    const int wr_r = tid >> 2;
    const int wr_c = (tid & 3) * 8;
    const _Float16* wsrc = W + (size_t)(n0 + wr_r) * EMBED + wr_c;
    const int ldst_off = wr_r * 40 + wr_c;

    const v8f vzero = {0.f, 0.f, 0.f, 0.f, 0.f, 0.f, 0.f, 0.f};
    v8f acc[2][4] = {{vzero, vzero, vzero, vzero}, {vzero, vzero, vzero, vzero}};

    // prologue: fill buffer 0
    async_b128(wsrc, &ldsW[0][ldst_off]);

    int cur = 0;
    for (int k0 = 0; k0 < EMBED; k0 += 32) {
        wait_async0();       // this wave's pending tile is in LDS
        __syncthreads();     // all waves' pieces are in LDS
        if (k0 + 32 < EMBED) // prefetch next tile into the other buffer
            async_b128(wsrc + k0 + 32, &ldsW[cur ^ 1][ldst_off]);

        // A fragments (two 16-row M-tiles) straight from global
        FragA a[2];
#pragma unroll
        for (int mt = 0; mt < 2; ++mt) {
            const _Float16* xr = X + (size_t)(m0 + mt * 16 + ln) * EMBED + k0;
            a[mt].q[0] = *(const uint4*)(xr + hi * 8);
            a[mt].q[1] = *(const uint4*)(xr + 16 + hi * 8);
        }

#pragma unroll
        for (int nt = 0; nt < 4; ++nt) {
            FragA b;  // lane ln = column n, K = hi*8..+7 then 16+hi*8..+7
            const _Float16* wrp = &ldsW[cur][(nt * 16 + ln) * 40 + hi * 8];
            b.q[0] = *(const uint4*)(wrp);
            b.q[1] = *(const uint4*)(wrp + 16);
            acc[0][nt] = wmma_f16(a[0], b, acc[0][nt]);
            acc[1][nt] = wmma_f16(a[1], b, acc[1][nt]);
        }
        cur ^= 1;
    }

    // epilogue: C/D layout -> element (M = hi*8+r, N = ln) per VGPR r
#pragma unroll
    for (int mt = 0; mt < 2; ++mt) {
#pragma unroll
        for (int nt = 0; nt < 4; ++nt) {
#pragma unroll
            for (int r = 0; r < 8; ++r) {
                int row = m0 + mt * 16 + hi * 8 + r;  // global M
                int col = n0 + nt * 16 + ln;          // global N
                float val = acc[mt][nt][r] + bias[col];
                if (mode == 2) {
                    ((float*)Y)[(size_t)row * EMBED + col] = val;
                } else {
                    int b = row >> 11, s = row & (SEQ - 1);
                    int h = col >> 6,  d = col & (HDIM - 1);
                    size_t bh = (size_t)(b * HEADS + h);
                    if (mode == 0)
                        ((_Float16*)Y)[(bh * SEQ + s) * HDIM + d] = (_Float16)val;
                    else
                        ((_Float16*)Y)[(bh * HDIM + d) * SEQ + s] = (_Float16)val;
                }
            }
        }
    }
}

// ---------------------------------------------------------------------------
// Flash-style attention. One wave per 16-query tile; loop over 32-key blocks.
// Q,K: (BH, S, D) f16.  Vt: (BH, D, S) f16.  O: (B, S, E) f16.
// Mask is the reference's faithful "+1.0 outside band" additive mask.
// ---------------------------------------------------------------------------
__global__ __launch_bounds__(256) void attn_wmma(
    const _Float16* __restrict__ Q,
    const _Float16* __restrict__ K,
    const _Float16* __restrict__ Vt,
    _Float16* __restrict__ O) {
    __shared__ _Float16 ldsP[8][16 * 40];  // per-wave 16x32 P tile, stride 40

    const int tid  = threadIdx.x;
    const int wave = tid >> 5;
    const int lane = tid & 31;
    const int ln   = lane & 15;
    const int hi   = lane >> 4;

    const int wid = blockIdx.x * 8 + wave;
    const int bh  = wid >> 7;        // 32 (b,h) pairs
    const int qt  = wid & 127;       // 128 query tiles of 16
    const int q0  = qt * 16;

    const _Float16* Qb = Q  + (size_t)bh * SEQ * HDIM;
    const _Float16* Kb = K  + (size_t)bh * SEQ * HDIM;
    const _Float16* Vb = Vt + (size_t)bh * HDIM * SEQ;

    // Q fragments: K-dim (head dim) chunks [0,32) and [32,64)
    FragA aq[2];
    {
        const _Float16* qr = Qb + (size_t)(q0 + ln) * HDIM;
        aq[0].q[0] = *(const uint4*)(qr + hi * 8);
        aq[0].q[1] = *(const uint4*)(qr + 16 + hi * 8);
        aq[1].q[0] = *(const uint4*)(qr + 32 + hi * 8);
        aq[1].q[1] = *(const uint4*)(qr + 48 + hi * 8);
    }

    const v8f vzero = {0.f, 0.f, 0.f, 0.f, 0.f, 0.f, 0.f, 0.f};
    v8f o_acc[4] = {vzero, vzero, vzero, vzero};
    float mrow[8], lrow[8];
#pragma unroll
    for (int r = 0; r < 8; ++r) { mrow[r] = -1e30f; lrow[r] = 0.f; }

    const float scale = 0.125f;  // 1/sqrt(64)
    _Float16* myP = ldsP[wave];

    for (int kt = 0; kt < SEQ; kt += 32) {
        // prefetch next key/value blocks into cache (global_prefetch_b8)
        if (kt + 32 < SEQ) {
            __builtin_prefetch(Kb + (size_t)(kt + 32 + lane) * HDIM, 0, 3);
            __builtin_prefetch(Vb + (size_t)(lane * 2) * SEQ + kt + 32, 0, 3);
        }

        // --- scores: 16 x 32 = two 16x16 WMMA tiles, K=64 chained ---
        v8f sc[2];
#pragma unroll
        for (int nt = 0; nt < 2; ++nt) {
            FragA bk0, bk1;  // B[k=d][n=key]: lane ln = key, d in table order
            const _Float16* kr = Kb + (size_t)(kt + nt * 16 + ln) * HDIM + hi * 8;
            bk0.q[0] = *(const uint4*)(kr);
            bk0.q[1] = *(const uint4*)(kr + 16);
            bk1.q[0] = *(const uint4*)(kr + 32);
            bk1.q[1] = *(const uint4*)(kr + 48);
            v8f c = vzero;
            c = wmma_f16(aq[0], bk0, c);
            c = wmma_f16(aq[1], bk1, c);
            sc[nt] = c;
        }

        // --- scale + additive band mask + online softmax ---
        float alpha[8];
#pragma unroll
        for (int r = 0; r < 8; ++r) {
            int qi = q0 + hi * 8 + r;
            int d0 = qi - (kt + ln);        if (d0 < 0) d0 = -d0;
            int d1 = qi - (kt + 16 + ln);   if (d1 < 0) d1 = -d1;
            float s0 = sc[0][r] * scale + ((d0 <= HALF_WIN) ? 0.f : 1.f);
            float s1 = sc[1][r] * scale + ((d1 <= HALF_WIN) ? 0.f : 1.f);
            float mx = fmaxf(s0, s1);
            mx = fmaxf(mx, __shfl_xor(mx, 1));
            mx = fmaxf(mx, __shfl_xor(mx, 2));
            mx = fmaxf(mx, __shfl_xor(mx, 4));
            mx = fmaxf(mx, __shfl_xor(mx, 8));
            float mnew = fmaxf(mrow[r], mx);
            alpha[r] = __expf(mrow[r] - mnew);
            mrow[r] = mnew;
            float p0 = __expf(s0 - mnew);
            float p1 = __expf(s1 - mnew);
            sc[0][r] = p0;
            sc[1][r] = p1;
            float rs = p0 + p1;
            rs += __shfl_xor(rs, 1);
            rs += __shfl_xor(rs, 2);
            rs += __shfl_xor(rs, 4);
            rs += __shfl_xor(rs, 8);
            lrow[r] = lrow[r] * alpha[r] + rs;
        }

        // rescale output accumulators
#pragma unroll
        for (int nt = 0; nt < 4; ++nt)
#pragma unroll
            for (int r = 0; r < 8; ++r) o_acc[nt][r] *= alpha[r];

        // --- transpose P from C/D layout to A layout via per-wave LDS ---
#pragma unroll
        for (int r = 0; r < 8; ++r) {
            int row = hi * 8 + r;
            myP[row * 40 + ln]      = (_Float16)sc[0][r];
            myP[row * 40 + 16 + ln] = (_Float16)sc[1][r];
        }
        __syncthreads();
        FragA ap;
        {
            const _Float16* pr = myP + ln * 40;
            ap.q[0] = *(const uint4*)(pr + hi * 8);
            ap.q[1] = *(const uint4*)(pr + 16 + hi * 8);
        }

        // --- O += P @ V : 4 WMMAs over d-tiles, B from transposed V ---
#pragma unroll
        for (int nt = 0; nt < 4; ++nt) {
            FragA bv;  // B[k=key][n=d]: lane ln = d column, keys in table order
            const _Float16* vr = Vb + (size_t)(nt * 16 + ln) * SEQ + kt + hi * 8;
            bv.q[0] = *(const uint4*)(vr);
            bv.q[1] = *(const uint4*)(vr + 16);
            o_acc[nt] = wmma_f16(ap, bv, o_acc[nt]);
        }
        __syncthreads();
    }

    // epilogue: normalize and scatter to (B, S, E) f16
    const int b = bh >> 4, h = bh & 15;
#pragma unroll
    for (int nt = 0; nt < 4; ++nt) {
#pragma unroll
        for (int r = 0; r < 8; ++r) {
            int s = q0 + hi * 8 + r;
            int e = h * HDIM + nt * 16 + ln;
            O[((size_t)(b * SEQ + s)) * EMBED + e] = (_Float16)(o_acc[nt][r] / lrow[r]);
        }
    }
}

// ---------------------------------------------------------------------------
// Launcher
// ---------------------------------------------------------------------------
extern "C" void kernel_launch(void* const* d_in, const int* in_sizes, int n_in,
                              void* d_out, int out_size, void* d_ws, size_t ws_size,
                              hipStream_t stream) {
    const float* q_in = (const float*)d_in[0];
    const float* k_in = (const float*)d_in[1];
    const float* v_in = (const float*)d_in[2];
    const float* Wq = (const float*)d_in[3];
    const float* bq = (const float*)d_in[4];
    const float* Wk = (const float*)d_in[5];
    const float* bk = (const float*)d_in[6];
    const float* Wv = (const float*)d_in[7];
    const float* bv = (const float*)d_in[8];
    const float* Wo = (const float*)d_in[9];
    const float* bo = (const float*)d_in[10];

    char* ws = (char*)d_ws;
    const size_t NX = (size_t)MROWS * EMBED;   // 4M elements
    const size_t NW = (size_t)EMBED * EMBED;   // 1M elements

    _Float16* xq  = (_Float16*)(ws);                       // 8 MB
    _Float16* xk  = (_Float16*)(ws + 8u * 1024 * 1024);
    _Float16* xv  = (_Float16*)(ws + 16u * 1024 * 1024);
    _Float16* wqh = (_Float16*)(ws + 24u * 1024 * 1024);   // 2 MB each
    _Float16* wkh = (_Float16*)(ws + 26u * 1024 * 1024);
    _Float16* wvh = (_Float16*)(ws + 28u * 1024 * 1024);
    _Float16* woh = (_Float16*)(ws + 30u * 1024 * 1024);
    _Float16* qw  = (_Float16*)(ws + 32u * 1024 * 1024);   // (bh,s,d)
    _Float16* kw  = (_Float16*)(ws + 40u * 1024 * 1024);   // (bh,s,d)
    _Float16* vtw = (_Float16*)(ws + 48u * 1024 * 1024);   // (bh,d,s)
    _Float16* att = (_Float16*)(ws + 56u * 1024 * 1024);   // (b,s,e)

    const int TB = 256;
    const int gx4 = (int)((NX / 4 + TB - 1) / TB);
    const int gw4 = (int)((NW / 4 + TB - 1) / TB);
    cvt_f32_to_f16<<<gx4, TB, 0, stream>>>((const float4*)q_in, (uint2*)xq, (int)(NX / 4));
    cvt_f32_to_f16<<<gx4, TB, 0, stream>>>((const float4*)k_in, (uint2*)xk, (int)(NX / 4));
    cvt_f32_to_f16<<<gx4, TB, 0, stream>>>((const float4*)v_in, (uint2*)xv, (int)(NX / 4));
    cvt_f32_to_f16<<<gw4, TB, 0, stream>>>((const float4*)Wq, (uint2*)wqh, (int)(NW / 4));
    cvt_f32_to_f16<<<gw4, TB, 0, stream>>>((const float4*)Wk, (uint2*)wkh, (int)(NW / 4));
    cvt_f32_to_f16<<<gw4, TB, 0, stream>>>((const float4*)Wv, (uint2*)wvh, (int)(NW / 4));
    cvt_f32_to_f16<<<gw4, TB, 0, stream>>>((const float4*)Wo, (uint2*)woh, (int)(NW / 4));

    dim3 ggrid(MROWS / 256, EMBED / 64);  // 16 x 16
    gemm_wmma<<<ggrid, 256, 0, stream>>>(xq, wqh, bq, (void*)qw, 0);
    gemm_wmma<<<ggrid, 256, 0, stream>>>(xk, wkh, bk, (void*)kw, 0);
    gemm_wmma<<<ggrid, 256, 0, stream>>>(xv, wvh, bv, (void*)vtw, 1);

    attn_wmma<<<(BATCH * HEADS * (SEQ / 16)) / 8, 256, 0, stream>>>(qw, kw, vtw, att);

    gemm_wmma<<<ggrid, 256, 0, stream>>>(att, woh, bo, d_out, 2);
}
